// ALIKEDLossWrapper_36060545417843
// MI455X (gfx1250) — matrix operand order repaired and checked by reading, not compiled, not run on hardware
//
#include <hip/hip_runtime.h>

#define BATCH 8
#define NPTS  2048
#define DDIM  128
#define IMGSZ 768
#define EPSV  1e-8f

#define LOG2E    1.44269504088896340f
#define LN2      0.69314718055994531f
#define TEN_L2E  14.4269504088896340f   // 10 * log2(e)

typedef __attribute__((ext_vector_type(16))) _Float16 v16h;
typedef __attribute__((ext_vector_type(2)))  __fp16   v2fp16;  // cvt_pkrtz return type
typedef __attribute__((ext_vector_type(8)))  float    v8f;

// accum layout (floats):
// 0: cnt   1: rp_sum   2: ds_sum   3: re_num   4: s_sum
// 5: dispA_sum 6: dispA_cnt 7: dispB_sum 8: dispB_cnt

__device__ __forceinline__ float wave_sum(float v) {
  #pragma unroll
  for (int m = 1; m < 32; m <<= 1) v += __shfl_xor(v, m, 32);
  return v;
}

__global__ void init_accum(float* accum) {
  if (threadIdx.x < 16) accum[threadIdx.x] = 0.0f;
}

// -------- Stage 1: projective warp (A->B with H, B->A with inv(H)) ----------
__global__ void warp_kernel(const float* __restrict__ kpts_a,
                            const float* __restrict__ kpts_b,
                            const float* __restrict__ H,
                            float* __restrict__ ka_w, float* __restrict__ kb_w) {
  int id = blockIdx.x * blockDim.x + threadIdx.x;
  if (id >= BATCH * NPTS) return;
  int b = id / NPTS;
  const float* h = H + b * 9;
  float h0=h[0],h1=h[1],h2=h[2],h3=h[3],h4=h[4],h5=h[5],h6=h[6],h7=h[7],h8=h[8];
  float c00 =  (h4*h8 - h5*h7);
  float c01 = -(h3*h8 - h5*h6);
  float c02 =  (h3*h7 - h4*h6);
  float det = h0*c00 + h1*c01 + h2*c02;
  float inv = 1.0f / det;
  float i0 =  c00*inv;
  float i1 = -(h1*h8 - h2*h7)*inv;
  float i2 =  (h1*h5 - h2*h4)*inv;
  float i3 =  c01*inv;
  float i4 =  (h0*h8 - h2*h6)*inv;
  float i5 = -(h0*h5 - h2*h3)*inv;
  float i6 =  c02*inv;
  float i7 = -(h0*h7 - h1*h6)*inv;
  float i8 =  (h0*h4 - h1*h3)*inv;

  float ax = kpts_a[id*2], ay = kpts_a[id*2+1];
  float px = h0*ax + h1*ay + h2;
  float py = h3*ax + h4*ay + h5;
  float pz = h6*ax + h7*ay + h8;
  ka_w[id*2]   = px / (pz + EPSV);
  ka_w[id*2+1] = py / (pz + EPSV);

  float bx = kpts_b[id*2], by = kpts_b[id*2+1];
  float qx = i0*bx + i1*by + i2;
  float qy = i3*bx + i4*by + i5;
  float qz = i6*bx + i7*by + i8;
  kb_w[id*2]   = qx / (qz + EPSV);
  kb_w[id*2+1] = qy / (qz + EPSV);
}

// -------- Stage 2: fused NN matching + reprojection loss --------------------
__global__ __launch_bounds__(256) void match_kernel(
    const float* __restrict__ kpts_a, const float* __restrict__ kpts_b,
    const float* __restrict__ ka_w,  const float* __restrict__ kb_w,
    const float* __restrict__ scores_a,
    int* __restrict__ idx_b, float* __restrict__ valid, float* accum) {
  __shared__ float2 sb[NPTS];                    // kpts_b for this batch (16 KB)
  int b    = blockIdx.x / (NPTS / 256);
  int nblk = blockIdx.x % (NPTS / 256);
  const float2* kb = (const float2*)(kpts_b + (size_t)b * NPTS * 2);
  for (int i = threadIdx.x; i < NPTS; i += 256) sb[i] = kb[i];
  __syncthreads();

  int n = nblk * 256 + threadIdx.x;
  int g = b * NPTS + n;
  float px = ka_w[g*2], py = ka_w[g*2+1];
  float best = 3.4e38f; int bi = 0;
  #pragma unroll 4
  for (int m = 0; m < NPTS; ++m) {
    float2 q = sb[m];
    float dx = px - q.x, dy = py - q.y;
    float d2 = dx*dx + dy*dy;
    if (d2 < best) { best = d2; bi = m; }
  }
  idx_b[g] = bi;
  float v = (sqrtf(best) < 5.0f) ? 1.0f : 0.0f;
  valid[g] = v;

  float2 s2 = sb[bi];
  float e_ab = sqrtf((px-s2.x)*(px-s2.x) + (py-s2.y)*(py-s2.y) + 1e-12f);
  float axp = kpts_a[g*2], ayp = kpts_a[g*2+1];
  float wx = kb_w[(size_t)(b*NPTS + bi)*2], wy = kb_w[(size_t)(b*NPTS + bi)*2 + 1];
  float e_ba = sqrtf((axp-wx)*(axp-wx) + (ayp-wy)*(ayp-wy) + 1e-12f);

  float rcnt = wave_sum(v);
  float rrp  = wave_sum(0.5f * (e_ab + e_ba) * v);
  float rsc  = wave_sum(scores_a[g] * v);
  if ((threadIdx.x & 31) == 0) {
    atomicAdd(&accum[0], rcnt);
    atomicAdd(&accum[1], rrp);
    atomicAdd(&accum[4], rsc);
  }
}

// -------- Stage 3: dispersity (peaky) loss ----------------------------------
__global__ __launch_bounds__(256) void disp_kernel(
    const float* __restrict__ smap, const float* __restrict__ kpts,
    float* accum, int sum_idx) {
  int id = blockIdx.x * blockDim.x + threadIdx.x;   // grid exactly covers B*N
  int b = id / NPTS;
  float kx = kpts[id*2], ky = kpts[id*2+1];
  int x = (int)floorf(kx), y = (int)floorf(ky);
  float v = (x >= 2 && x < IMGSZ-2 && y >= 2 && y < IMGSZ-2) ? 1.0f : 0.0f;
  int xc = min(max(x, 2), IMGSZ - 3);
  int yc = min(max(y, 2), IMGSZ - 3);
  const float* base = smap + (size_t)b * IMGSZ * IMGSZ;
  float p[25];
  float mx = -3.4e38f;
  #pragma unroll
  for (int j = 0; j < 25; ++j) {
    int yy = yc + j / 5 - 2, xx = xc + j % 5 - 2;
    p[j] = base[yy * IMGSZ + xx];
    mx = fmaxf(mx, p[j]);
  }
  float sum = 0.f, wsum = 0.f;
  #pragma unroll
  for (int j = 0; j < 25; ++j) {
    float e = __builtin_amdgcn_exp2f((p[j] - mx) * LOG2E);
    sum += e;
    float gy = (float)(j / 5 - 2), gx = (float)(j % 5 - 2);
    wsum += e * sqrtf(gx*gx + gy*gy);
  }
  float disp = wsum / sum;
  float rsum = wave_sum(disp * v);
  float rcnt = wave_sum(v);
  if ((threadIdx.x & 31) == 0) {
    atomicAdd(&accum[sum_idx], rsum);
    atomicAdd(&accum[sum_idx + 1], rcnt);
  }
}

// -------- Stage 4: fused descriptor similarity (WMMA) + NRE/reliability -----
// Block: 256 threads = 8 waves; each wave owns a 16-row A tile; block covers
// 128 rows and sweeps all 2048 columns in chunks of 16. sim never hits HBM.
__global__ __launch_bounds__(256) void sim_wmma_kernel(
    const float* __restrict__ desc_a, const float* __restrict__ desc_b,
    const float* __restrict__ scores_a,
    const int* __restrict__ idx_b, const float* __restrict__ valid,
    float* accum) {
  __shared__ __align__(32) _Float16 bt[DDIM * 16];  // B^T tile [k][n], 4 KB
  __shared__ float rowds[128];
  __shared__ float rowre[128];
  __shared__ float rowsel[128];

  const int tid  = threadIdx.x;
  const int wave = tid >> 5;
  const int lane = tid & 31;
  const int batch  = blockIdx.y;
  const int rowblk = blockIdx.x;                       // 0..15
  const int gr0 = batch * NPTS + rowblk * 128 + wave * 16;  // flat row base

  // A fragment: ISA 16-bit A layout. lane%16 = M row; halves hold K with the
  // +8 K-offset for lanes 16..31. K base per v16h chunk kc = kc*32.
  v16h areg[4];
  const int arow = gr0 + (lane & 15);
  const int khi  = (lane >> 4) << 3;
  #pragma unroll
  for (int kc = 0; kc < 4; ++kc) {
    #pragma unroll
    for (int h = 0; h < 16; ++h) {
      int K = kc * 32 + (h & 7) + ((h >> 3) << 4) + khi;
      areg[kc][h] = (_Float16)desc_a[(size_t)arow * DDIM + K];
    }
  }

  int myidx[8];
  #pragma unroll
  for (int e = 0; e < 8; ++e) myidx[e] = idx_b[gr0 + e + khi];

  float pds[8], pre[8], psel[8];
  #pragma unroll
  for (int e = 0; e < 8; ++e) { pds[e] = 0.f; pre[e] = 0.f; psel[e] = 0.f; }

  const float* db = desc_b + (size_t)batch * NPTS * DDIM;
  unsigned* bt32 = (unsigned*)bt;        // word w = k*8 + npair  (n = 2*npair)
  const int lcol = lane & 15;

  for (int mc = 0; mc < NPTS; mc += 16) {
    __syncthreads();
    // stage B^T tile: bt[k][n] = desc_b[mc+n][k] as f16.
    // Each thread emits LDS-contiguous half-pairs via packed convert.
    #pragma unroll
    for (int j = 0; j < 4; ++j) {
      int w  = tid + 256 * j;            // 0..1023
      int k  = w & 127;                  // consecutive lanes -> consecutive k
      int np = w >> 7;                   // 0..7
      int n0 = np * 2;
      float x0 = db[(size_t)(mc + n0)     * DDIM + k];
      float x1 = db[(size_t)(mc + n0 + 1) * DDIM + k];
      union { v2fp16 h; unsigned u; } cvt;
      cvt.h = __builtin_amdgcn_cvt_pkrtz(x0, x1);
      bt32[k * 8 + np] = cvt.u;
    }
    __syncthreads();

    v8f c = {};
    #pragma unroll
    for (int kc = 0; kc < 4; ++kc) {
      // B fragment: lane%16 = N column, 16 halves = K in the same order as A
      v16h bv;
      #pragma unroll
      for (int h = 0; h < 16; ++h)
        bv[h] = bt[(kc * 32 + (h & 7) + ((h >> 3) << 4) + khi) * 16 + lcol];
      c = __builtin_amdgcn_wmma_f32_16x16x32_f16(false, areg[kc], false, bv,
                                                 (short)0, c, false, false);
    }

    int col = mc + lcol;
    #pragma unroll
    for (int e = 0; e < 8; ++e) {
      float s = c[e];                                        // sim[e+khi][col]
      pds[e] += __builtin_amdgcn_exp2f(__builtin_fmaf(s, TEN_L2E, -TEN_L2E));
      pre[e] += __builtin_amdgcn_exp2f(s * LOG2E);
      psel[e] = (myidx[e] == col) ? s : psel[e];             // branch-free capture
    }
  }

  // Sum partials across the 16 lanes that share each output row.
  #pragma unroll
  for (int e = 0; e < 8; ++e) {
    #pragma unroll
    for (int m = 1; m < 16; m <<= 1) {
      pds[e]  += __shfl_xor(pds[e],  m, 32);
      pre[e]  += __shfl_xor(pre[e],  m, 32);
      psel[e] += __shfl_xor(psel[e], m, 32);  // exactly one lane nonzero
    }
  }
  if ((lane & 15) == 0) {
    #pragma unroll
    for (int e = 0; e < 8; ++e) {
      rowds[wave * 16 + e + khi]  = pds[e];
      rowre[wave * 16 + e + khi]  = pre[e];
      rowsel[wave * 16 + e + khi] = psel[e];
    }
  }
  __syncthreads();

  if (tid < 128) {                       // waves 0..3, fully active
    int g = batch * NPTS + rowblk * 128 + tid;
    float v  = valid[g];
    float sa = scores_a[g];
    float s  = rowsel[tid];
    float q  = __builtin_amdgcn_exp2f(__builtin_fmaf(s, TEN_L2E, -TEN_L2E)) / rowds[tid];
    float r  = __builtin_amdgcn_exp2f(s * LOG2E) / rowre[tid];
    float cds = -(__builtin_amdgcn_logf(q + EPSV) * LN2) * v;
    float cre = (1.0f - r) * sa * v;
    cds = wave_sum(cds);
    cre = wave_sum(cre);
    if ((tid & 31) == 0) {
      atomicAdd(&accum[2], cds);
      atomicAdd(&accum[3], cre);
    }
  }
}

// -------- Stage 5: combine ---------------------------------------------------
__global__ void finalize_kernel(const float* accum, float* out) {
  float cnt = accum[0];
  float loss_rp = cnt > 0.f ? accum[1] / fmaxf(cnt, 1.f) : 0.f;
  float dA = accum[6] > 0.f ? accum[5] / fmaxf(accum[6], 1.f) : 0.f;
  float dB = accum[8] > 0.f ? accum[7] / fmaxf(accum[8], 1.f) : 0.f;
  float loss_pk = 0.5f * (dA + dB);
  float loss_ds = cnt > 0.f ? accum[2] / fmaxf(cnt, 1.f) : 0.f;
  float loss_re = accum[4] > 0.f ? accum[3] / fmaxf(accum[4], EPSV) : 0.f;
  float total = 1.0f * loss_rp + 0.5f * loss_pk + 5.0f * loss_ds + 1.0f * loss_re;
  out[0] = loss_rp; out[1] = loss_pk; out[2] = loss_ds; out[3] = loss_re; out[4] = total;
}

extern "C" void kernel_launch(void* const* d_in, const int* in_sizes, int n_in,
                              void* d_out, int out_size, void* d_ws, size_t ws_size,
                              hipStream_t stream) {
  (void)in_sizes; (void)n_in; (void)out_size; (void)ws_size;
  const float* kpts_a   = (const float*)d_in[0];
  const float* desc_a   = (const float*)d_in[1];
  const float* scores_a = (const float*)d_in[2];
  const float* smap_a   = (const float*)d_in[3];
  const float* kpts_b   = (const float*)d_in[4];
  const float* desc_b   = (const float*)d_in[5];
  const float* smap_b   = (const float*)d_in[7];
  const float* H_ab     = (const float*)d_in[8];

  float* ka_w  = (float*)d_ws;                     // B*N*2
  float* kb_w  = ka_w + BATCH * NPTS * 2;          // B*N*2
  int*   idxb  = (int*)(kb_w + BATCH * NPTS * 2);  // B*N
  float* valid = (float*)(idxb + BATCH * NPTS);    // B*N
  float* accum = valid + BATCH * NPTS;             // 16

  init_accum<<<1, 32, 0, stream>>>(accum);
  warp_kernel<<<(BATCH * NPTS + 255) / 256, 256, 0, stream>>>(
      kpts_a, kpts_b, H_ab, ka_w, kb_w);
  match_kernel<<<BATCH * (NPTS / 256), 256, 0, stream>>>(
      kpts_a, kpts_b, ka_w, kb_w, scores_a, idxb, valid, accum);
  disp_kernel<<<(BATCH * NPTS + 255) / 256, 256, 0, stream>>>(smap_a, kpts_a, accum, 5);
  disp_kernel<<<(BATCH * NPTS + 255) / 256, 256, 0, stream>>>(smap_b, kpts_b, accum, 7);
  dim3 g(NPTS / 128, BATCH);
  sim_wmma_kernel<<<g, 256, 0, stream>>>(desc_a, desc_b, scores_a, idxb, valid, accum);
  finalize_kernel<<<1, 1, 0, stream>>>(accum, (float*)d_out);
}
